// SimCSELoss_39865886442246
// MI455X (gfx1250) — matrix operand order repaired and checked by reading, not compiled, not run on hardware
//
#include <hip/hip_runtime.h>
#include <hip/hip_bf16.h>

typedef __attribute__((ext_vector_type(16))) _Float16 v16h;
typedef __attribute__((ext_vector_type(8)))  _Float16 v8h;
typedef __attribute__((ext_vector_type(8)))  float    v8f;

#define N_ROWS   8192
#define DIM      768
#define NTILES   (N_ROWS / 16)   // 512 column tiles of 16
#define KCHUNKS  (DIM / 32)      // 24 WMMA steps of K=32
#define TEMP_INV 2.0f            // 1 / 0.5
#define EPS_F    1e-8f
#define LDA      776             // padded LDS row stride (halves): 1552B = 388 dw, 388%64=4 -> conflict-free

// ---------------------------------------------------------------------------
// Kernel 1: per-row L2 normalize, emit fp16 Xn (row-major [8192 x 768]).
// ---------------------------------------------------------------------------
__global__ __launch_bounds__(256)
void k_normalize(const float* __restrict__ x, _Float16* __restrict__ xn) {
    const int row = blockIdx.x;
    const int tid = threadIdx.x;            // 0..255
    const float* xr = x + (size_t)row * DIM;

    float v0 = xr[tid];
    float v1 = xr[tid + 256];
    float v2 = xr[tid + 512];
    float ss = v0 * v0 + v1 * v1 + v2 * v2;

    #pragma unroll
    for (int m = 1; m < 32; m <<= 1) ss += __shfl_xor(ss, m, 32);

    __shared__ float wsum[8];
    if ((tid & 31) == 0) wsum[tid >> 5] = ss;
    __syncthreads();

    float tot = 0.f;
    #pragma unroll
    for (int i = 0; i < 8; ++i) tot += wsum[i];

    const float inv = 1.0f / fmaxf(sqrtf(tot), EPS_F);   // clamp like reference

    _Float16* xo = xn + (size_t)row * DIM;
    xo[tid]       = (_Float16)(v0 * inv);
    xo[tid + 256] = (_Float16)(v1 * inv);
    xo[tid + 512] = (_Float16)(v2 * inv);
}

// ---------------------------------------------------------------------------
// Kernel 2: per 16-row strip, stream 512 column tiles through
// v_wmma_f32_16x16x32_f16; accumulate sum_j exp(2*cos) per row (diag masked),
// capture target logit S[i, i^1], emit per-block partial of (log(sum)-target).
// ---------------------------------------------------------------------------
__global__ __launch_bounds__(256)
void k_simcse_gemm(const _Float16* __restrict__ xn,
                   float* __restrict__ partials) {
    __shared__ __align__(32) _Float16 lA[16 * LDA];
    __shared__ float rs[16];     // per-row sum of exp
    __shared__ float tgt[16];    // per-row target logit
    __shared__ float term[16];

    const int tid  = threadIdx.x;      // 0..255
    const int lane = tid & 31;
    const int wave = tid >> 5;         // 0..7
    const int it   = blockIdx.x;       // row tile index
    const int i0   = it * 16;

    if (tid < 16) { rs[tid] = 0.f; tgt[tid] = 0.f; }

    // cooperative A strip load: 16 rows x 768 halves -> padded LDS
    for (int c = tid; c < 16 * (DIM / 8); c += 256) {
        const int r   = c / (DIM / 8);
        const int col = (c % (DIM / 8)) * 8;
        *(v8h*)(&lA[r * LDA + col]) =
            *(const v8h*)(xn + (size_t)(i0 + r) * DIM + col);
    }
    __syncthreads();

    const int half = lane >> 4;        // 0: lanes 0-15, 1: lanes 16-31
    const int l16  = lane & 15;

    // A-fragment base (contiguous K order k = 16*half + s, same as B):
    // lanes 0-15: row l16, K = {0..15}; lanes 16-31: K = {16..31}
    const _Float16* aBase = &lA[l16 * LDA + half * 16];

    v8f acc = {};   // per-lane running sum of exp for rows M = r (+8)

    for (int jt = wave; jt < NTILES; jt += 8) {
        // B fragment (K x 16): lane = column l16; half-wave selects K 0-15 / 16-31
        const _Float16* bBase = xn + (size_t)(jt * 16 + l16) * DIM + half * 16;
        if (jt + 8 < NTILES)  // pull next tile toward WGP caches
            __builtin_prefetch(xn + (size_t)((jt + 8) * 16 + l16) * DIM, 0, 3);

        // Opaque per-iteration offset: defeats LICM hoisting of the 24 A
        // fragments (which spilled 192 VGPRs to scratch). ds_load path is kept
        // because the base still traces to the __shared__ array.
        int aoff = 0;
        asm volatile("" : "+s"(aoff));
        const _Float16* aB = aBase + aoff;

        v8f c = {};
        v16h b_cur = *(const v16h*)(bBase);          // K chunk 0
        #pragma unroll
        for (int kk = 0; kk < KCHUNKS; ++kk) {
            v16h b_nxt;
            if (kk + 1 < KCHUNKS)                    // software pipeline B
                b_nxt = *(const v16h*)(bBase + (kk + 1) * 32);
            v8h lo = *(const v8h*)(aB + kk * 32);        // ds_load_b128
            v8h hi = *(const v8h*)(aB + kk * 32 + 8);    // ds_load_b128
            v16h a;
            #pragma unroll
            for (int e = 0; e < 8; ++e) { a[e] = lo[e]; a[e + 8] = hi[e]; }
            c = __builtin_amdgcn_wmma_f32_16x16x32_f16(
                    false, a, false, b_cur, (short)0, c, false, false);
            if (kk + 1 < KCHUNKS) b_cur = b_nxt;
        }

        if (jt == it) {
            // diagonal tile: mask self-sim, capture target S[i, i^1]
            #pragma unroll
            for (int r = 0; r < 8; ++r) {
                const int M  = r + half * 8;    // local row
                const int Nn = l16;             // local col
                const float v = c[r] * TEMP_INV;
                float e = __expf(v);
                if (Nn == M) e = 0.f;           // diag mask (-1e12 -> exp ~ 0)
                if (Nn == (M ^ 1)) tgt[M] = v;  // unsup SimCSE positive
                acc[r] += e;
            }
        } else {
            #pragma unroll
            for (int r = 0; r < 8; ++r) acc[r] += __expf(c[r] * TEMP_INV);
        }
    }

    // reduce acc over the 16 lanes sharing each row, combine waves via LDS
    #pragma unroll
    for (int r = 0; r < 8; ++r) {
        float s = acc[r];
        s += __shfl_xor(s, 1, 16);
        s += __shfl_xor(s, 2, 16);
        s += __shfl_xor(s, 4, 16);
        s += __shfl_xor(s, 8, 16);
        if (l16 == r) atomicAdd(&rs[r + half * 8], s);   // ds_add_f32
    }
    __syncthreads();

    if (tid < 16) term[tid] = __logf(rs[tid]) - tgt[tid];
    __syncthreads();

    if (tid == 0) {
        float p = 0.f;
        #pragma unroll
        for (int i = 0; i < 16; ++i) p += term[i];
        partials[blockIdx.x] = p;    // sum of -logp over this strip's 16 rows
    }
}

// ---------------------------------------------------------------------------
// Kernel 3: reduce 512 block partials -> loss = sum / 8192
// ---------------------------------------------------------------------------
__global__ __launch_bounds__(256)
void k_final(const float* __restrict__ partials, float* __restrict__ out) {
    __shared__ float sm[8];
    const int tid = threadIdx.x;    // 256 threads
    float s = partials[tid] + partials[tid + 256];
    #pragma unroll
    for (int m = 1; m < 32; m <<= 1) s += __shfl_xor(s, m, 32);
    if ((tid & 31) == 0) sm[tid >> 5] = s;
    __syncthreads();
    if (tid == 0) {
        float t = 0.f;
        #pragma unroll
        for (int i = 0; i < 8; ++i) t += sm[i];
        out[0] = t / (float)N_ROWS;
    }
}

// ---------------------------------------------------------------------------
extern "C" void kernel_launch(void* const* d_in, const int* in_sizes, int n_in,
                              void* d_out, int out_size, void* d_ws, size_t ws_size,
                              hipStream_t stream) {
    const float* x = (const float*)d_in[0];
    _Float16* xn   = (_Float16*)d_ws;
    float* partials = (float*)((char*)d_ws + (size_t)N_ROWS * DIM * sizeof(_Float16));

    k_normalize  <<<N_ROWS, 256, 0, stream>>>(x, xn);
    k_simcse_gemm<<<NTILES, 256, 0, stream>>>(xn, partials);
    k_final      <<<1,      256, 0, stream>>>(partials, (float*)d_out);
}
